// ConvLSTMAutoencoder_53541062312022
// MI455X (gfx1250) — matrix operand order repaired and checked by reading, not compiled, not run on hardware
//
#include <hip/hip_runtime.h>
#include <hip/hip_bf16.h>
#include <stdint.h>

// ---------------------------------------------------------------------------
// ConvLSTM autoencoder for MI455X (gfx1250), WMMA f16 path.
//
// Per timestep per layer:
//   packa : im2col(cat(x_t, h_{t-1})) -> P[n][KPAD] f16 (WMMA-B-ready, L2-resident)
//   step  : z = Wg x P via v_wmma_f32_16x16x32_f16, fully unrolled over k-steps
//           (compile-time KSTEPS -> no register rotation, constant offsets),
//           fused branch-free LSTM gate math, cell state updated in place,
//           h_t written to the sequence buffer (doubles as next-step h input).
// Weights are pre-packed once per layer so one lane's 4 gate fragments per
// k-step are a single contiguous 128-byte block.
// ---------------------------------------------------------------------------

typedef __attribute__((ext_vector_type(16))) _Float16 v16h;
typedef __attribute__((ext_vector_type(8)))  float    v8f;

#define NPOS   8192     // B*H = 128*64 positions
#define TSTEPS 100
#define KPAD   256      // padded im2col depth (max real K = 192)

__device__ __forceinline__ float sigf(float x) {
    return 1.0f / (1.0f + __expf(-x));
}
// Branch-free tanh: 1 - 2/(e^{2x}+1); saturates correctly via __expf overflow.
__device__ __forceinline__ float tanh_fast(float x) {
    return 1.0f - 2.0f / (__expf(2.0f * x) + 1.0f);
}

// Zero-init a region of words (cell states + zero hidden buffers).
__global__ void zinit_kernel(uint32_t* __restrict__ p, int nwords) {
    int i = blockIdx.x * blockDim.x + threadIdx.x;
    if (i < nwords) p[i] = 0u;
}

// x [B,T,F] f32  ->  xh [T][B*64] f16 (time-major, n = b*64+f)
__global__ void xcvt_kernel(const float* __restrict__ x, _Float16* __restrict__ xh) {
    int i = blockIdx.x * blockDim.x + threadIdx.x;
    if (i >= 128 * TSTEPS * 64) return;
    int f = i & 63;
    int t = (i >> 6) % TSTEPS;
    int b = i / (TSTEPS * 64);
    xh[t * NPOS + b * 64 + f] = (_Float16)x[i];
}

// Pre-pack conv weights into WMMA A-fragment lane layout (once per layer).
// wp layout: [chtile][kstep][lane 0..31][gate 0..3][16 halves]
//   -> one lane's 4 gate fragments are 128B contiguous per k-step.
// A 16x32 f16, contiguous k order used here (B matches): lane<16 holds K 0..15 (+ks*32), lane>=16 +16.
__global__ void packw_kernel(const float* __restrict__ w, _Float16* __restrict__ wp,
                             int cinx, int chid) {
    int cintot = cinx + chid;
    int Ktot   = cintot * 3;
    int ksteps = (Ktot + 31) / 32;
    int chtiles = chid >> 4;
    int total = chtiles * ksteps * 32 * 4 * 16;
    int i = blockIdx.x * blockDim.x + threadIdx.x;
    if (i >= total) return;
    int e    = i & 15;
    int g    = (i >> 4) & 3;
    int lane = (i >> 6) & 31;
    int ks   = (i >> 11) % ksteps;
    int ct   = i / (ksteps << 11);
    int aoff = (lane < 16) ? 0 : 16;
    int k    = ks * 32 + aoff + e;
    float v = 0.0f;
    if (k < Ktot) {
        int ci = k / 3;
        int kh = k - ci * 3;
        int co = g * chid + ct * 16 + (lane & 15);
        v = w[((co * cintot + ci) * 3 + kh) * 3 + 1];   // center column kw=1
    }
    wp[i] = (_Float16)v;
}

// Build im2col operand P[n][KPAD] (f16) for one timestep.
// k = ci*3 + kh; row ci<cinx comes from xin, else from hprev; h-edge -> 0.
__global__ void packa_kernel(const _Float16* __restrict__ xin,
                             const _Float16* __restrict__ hprev,
                             _Float16* __restrict__ P, int cinx, int chid) {
    int i = blockIdx.x * blockDim.x + threadIdx.x;   // over NPOS*KPAD
    int n = i >> 8;
    int k = i & (KPAD - 1);
    int Ktot = (cinx + chid) * 3;
    _Float16 v = (_Float16)0.0f;
    if (k < Ktot) {
        int ci = k / 3;
        int kh = k - ci * 3;
        int hs = (n & 63) + kh - 1;
        if (hs >= 0 && hs < 64) {
            int sn = n + kh - 1;
            v = (ci < cinx) ? xin[ci * NPOS + sn] : hprev[(ci - cinx) * NPOS + sn];
        }
    }
    P[i] = v;
}

// One ConvLSTM timestep: fully unrolled WMMA GEMM + fused gate math.
// Wave -> (chtile, ntile): 512 n-tiles, chid/16 ch-tiles.
template <int KSTEPS>
__global__ __launch_bounds__(256) void lstm_step_kernel(
    const _Float16* __restrict__ P, const _Float16* __restrict__ wp,
    const float* __restrict__ bias, float* __restrict__ cst,
    _Float16* __restrict__ hout, int chid)
{
    const int lane   = threadIdx.x & 31;
    const int wave   = blockIdx.x * (blockDim.x >> 5) + (threadIdx.x >> 5);
    const int ntile  = wave & 511;
    const int chtile = wave >> 9;
    const int chbase = chtile * 16;

    const int n    = ntile * 16 + (lane & 15);   // output position (b*64 + h)
    const int boff = (lane < 16) ? 0 : 16;       // B fragment K-offset
    const int moff = (lane < 16) ? 0 : 8;        // C/D row offset

    // Bias init: one 32-byte vector load per gate (layout guarantees alignment).
    v8f acc[4];
#pragma unroll
    for (int g = 0; g < 4; ++g)
        acc[g] = *(const v8f*)(bias + g * chid + chbase + moff);

    const _Float16* prow = P + (size_t)n * KPAD + boff;
    const _Float16* wl   = wp + ((size_t)chtile * KSTEPS * 32 + lane) * 64;

#pragma unroll
    for (int ks = 0; ks < KSTEPS; ++ks) {
        v16h b = *(const v16h*)(prow + ks * 32);
#pragma unroll
        for (int g = 0; g < 4; ++g) {
            v16h a = *(const v16h*)(wl + (size_t)ks * 2048 + g * 16);
            acc[g] = __builtin_amdgcn_wmma_f32_16x16x32_f16(
                false, a, false, b, (short)0, acc[g], false, false);
        }
    }

    // Gate math: c' = sig(f)*c + sig(i)*tanh(g);  h' = sig(o)*tanh(c')
#pragma unroll
    for (int r = 0; r < 8; ++r) {
        const int ch  = chbase + r + moff;
        const int idx = ch * NPOS + n;
        float cold = cst[idx];
        float iv = sigf(acc[0][r]);
        float fv = sigf(acc[1][r]);
        float ov = sigf(acc[2][r]);
        float gv = tanh_fast(acc[3][r]);
        float c2 = fv * cold + iv * gv;
        float h2 = ov * tanh_fast(c2);
        cst[idx]  = c2;
        hout[idx] = (_Float16)h2;
    }
}

// Final 1x1 conv: out[b][t][f] = fc_b + sum_ch fc_w[ch] * dec2[t][ch][b*64+f]
__global__ void fconv_kernel(const _Float16* __restrict__ seq,
                             const float* __restrict__ fcw,
                             const float* __restrict__ fcb,
                             float* __restrict__ out)
{
    int i = blockIdx.x * blockDim.x + threadIdx.x;
    if (i >= 128 * TSTEPS * 64) return;
    int f = i & 63;
    int t = (i >> 6) % TSTEPS;
    int b = i / (TSTEPS * 64);
    int npos = b * 64 + f;
    float s = fcb[0];
#pragma unroll
    for (int ch = 0; ch < 16; ++ch)
        s += fcw[ch] * (float)seq[(t * 16 + ch) * NPOS + npos];
    out[i] = s;
}

extern "C" void kernel_launch(void* const* d_in, const int* in_sizes, int n_in,
                              void* d_out, int out_size, void* d_ws, size_t ws_size,
                              hipStream_t stream) {
    const float* x   = (const float*)d_in[0];
    const float* ew0 = (const float*)d_in[1];
    const float* eb0 = (const float*)d_in[2];
    const float* ew1 = (const float*)d_in[3];
    const float* eb1 = (const float*)d_in[4];
    const float* dw0 = (const float*)d_in[5];
    const float* db0 = (const float*)d_in[6];
    const float* dw1 = (const float*)d_in[7];
    const float* db1 = (const float*)d_in[8];
    const float* fcw = (const float*)d_in[9];
    const float* fcb = (const float*)d_in[10];

    char* ws = (char*)d_ws;
    size_t o = 0;
    // Zeroed region first (one zinit covers all four buffers).
    _Float16* zero16 = (_Float16*)(ws + o); o += (size_t)16 * NPOS * 2;   // 256 KB
    _Float16* zero32 = (_Float16*)(ws + o); o += (size_t)32 * NPOS * 2;   // 512 KB
    float*    c16    = (float*)   (ws + o); o += (size_t)16 * NPOS * 4;   // 512 KB
    float*    c32    = (float*)   (ws + o); o += (size_t)32 * NPOS * 4;   // 1 MB
    const int zero_words = (int)(o / 4);
    _Float16* Pbuf = (_Float16*)(ws + o); o += (size_t)NPOS * KPAD * 2;   // 4 MB (L2-resident)
    _Float16* wp0  = (_Float16*)(ws + o); o += 65536;                     // packed weights
    _Float16* wp1  = (_Float16*)(ws + o); o += 65536;
    _Float16* wp2  = (_Float16*)(ws + o); o += 65536;
    _Float16* wp3  = (_Float16*)(ws + o); o += 65536;
    _Float16* xh = (_Float16*)(ws + o); o += (size_t)TSTEPS * NPOS * 2;
    _Float16* s0 = (_Float16*)(ws + o); o += (size_t)TSTEPS * 16 * NPOS * 2;  // enc0 out
    _Float16* s1 = (_Float16*)(ws + o); o += (size_t)TSTEPS * 32 * NPOS * 2;  // enc1 out
    _Float16* s2 = (_Float16*)(ws + o); o += (size_t)TSTEPS * 32 * NPOS * 2;  // dec0 out
    _Float16* s3 = (_Float16*)(ws + o); o += (size_t)TSTEPS * 16 * NPOS * 2;  // dec1 out

    zinit_kernel<<<(zero_words + 255) / 256, 256, 0, stream>>>((uint32_t*)ws, zero_words);
    xcvt_kernel<<<(128 * TSTEPS * 64 + 255) / 256, 256, 0, stream>>>(x, xh);

    // Pack weights once per layer (tiny kernels).
    packw_kernel<<<16,  256, 0, stream>>>(ew0, wp0, 1,  16);  // 1*2*4*32*16  = 4096
    packw_kernel<<<80,  256, 0, stream>>>(ew1, wp1, 16, 32);  // 2*5*4*32*16  = 20480
    packw_kernel<<<96,  256, 0, stream>>>(dw0, wp2, 32, 32);  // 2*6*4*32*16  = 24576
    packw_kernel<<<40,  256, 0, stream>>>(dw1, wp3, 32, 16);  // 1*5*4*32*16  = 10240

    const int PACK_BLKS = (NPOS * KPAD) / 256;   // 8192 blocks

    // Encoder layer 0: Cin=1, Chid=16, ksteps=2 (512 waves -> 64 blocks)
    for (int t = 0; t < TSTEPS; ++t) {
        const _Float16* hp = t ? (s0 + (size_t)(t - 1) * 16 * NPOS) : zero16;
        packa_kernel<<<PACK_BLKS, 256, 0, stream>>>(xh + (size_t)t * NPOS, hp, Pbuf, 1, 16);
        lstm_step_kernel<2><<<64, 256, 0, stream>>>(Pbuf, wp0, eb0, c16,
                                                    s0 + (size_t)t * 16 * NPOS, 16);
    }
    // Encoder layer 1: Cin=16, Chid=32, ksteps=5 (1024 waves -> 128 blocks)
    for (int t = 0; t < TSTEPS; ++t) {
        const _Float16* hp = t ? (s1 + (size_t)(t - 1) * 32 * NPOS) : zero32;
        packa_kernel<<<PACK_BLKS, 256, 0, stream>>>(s0 + (size_t)t * 16 * NPOS, hp, Pbuf, 16, 32);
        lstm_step_kernel<5><<<128, 256, 0, stream>>>(Pbuf, wp1, eb1, c32,
                                                     s1 + (size_t)t * 32 * NPOS, 32);
    }
    // Decoder layer 0: Cin=32, Chid=32, ksteps=6; initial (h,c) = enc1 final.
    for (int t = 0; t < TSTEPS; ++t) {
        const _Float16* hp = t ? (s2 + (size_t)(t - 1) * 32 * NPOS)
                               : (s1 + (size_t)(TSTEPS - 1) * 32 * NPOS);
        packa_kernel<<<PACK_BLKS, 256, 0, stream>>>(s1 + (size_t)t * 32 * NPOS, hp, Pbuf, 32, 32);
        lstm_step_kernel<6><<<128, 256, 0, stream>>>(Pbuf, wp2, db0, c32,
                                                     s2 + (size_t)t * 32 * NPOS, 32);
    }
    // Decoder layer 1: Cin=32, Chid=16, ksteps=5; initial (h,c) = enc0 final.
    for (int t = 0; t < TSTEPS; ++t) {
        const _Float16* hp = t ? (s3 + (size_t)(t - 1) * 16 * NPOS)
                               : (s0 + (size_t)(TSTEPS - 1) * 16 * NPOS);
        packa_kernel<<<PACK_BLKS, 256, 0, stream>>>(s2 + (size_t)t * 32 * NPOS, hp, Pbuf, 32, 16);
        lstm_step_kernel<5><<<64, 256, 0, stream>>>(Pbuf, wp3, db1, c16,
                                                    s3 + (size_t)t * 16 * NPOS, 16);
    }
    // Final 1x1 conv 16 -> 1.
    fconv_kernel<<<(128 * TSTEPS * 64 + 255) / 256, 256, 0, stream>>>(
        s3, fcw, fcb, (float*)d_out);
}
